// EdgePredictor_38190849196540
// MI455X (gfx1250) — matrix operand order repaired and checked by reading, not compiled, not run on hardware
//
#include <hip/hip_runtime.h>
#include <hip/hip_bf16.h>
#include <cstdint>

#define NPTS 4096
#define HID  64
#define KNNK 15

typedef __attribute__((ext_vector_type(16))) _Float16 v16h;
typedef __attribute__((ext_vector_type(8)))  _Float16 v8h;
typedef __attribute__((ext_vector_type(8)))  float    v8f;
typedef __attribute__((ext_vector_type(4)))  unsigned u32x4;
typedef __attribute__((ext_vector_type(8)))  int      i32x8;
typedef __attribute__((ext_vector_type(4)))  int      i32x4;

// ---------- order-preserving float<->uint for atomicMax segment-max ----------
__device__ __forceinline__ unsigned enc_f32(float f) {
  unsigned u = __float_as_uint(f);
  return (u & 0x80000000u) ? ~u : (u | 0x80000000u);
}
__device__ __forceinline__ float dec_f32(unsigned u) {
  u = (u & 0x80000000u) ? (u ^ 0x80000000u) : ~u;
  return __uint_as_float(u);
}

// ---------------- KNN: one block per node, distances staged in LDS ----------
__global__ __launch_bounds__(256) void knn_kernel(const float* __restrict__ pts,
                                                  int* __restrict__ knn_idx) {
  __shared__ float d2[NPTS];
  __shared__ float rv[256];
  __shared__ int   ri[256];
  const int i  = blockIdx.x;
  const int t  = threadIdx.x;
  const float px = pts[3 * i + 0], py = pts[3 * i + 1], pz = pts[3 * i + 2];
  for (int j = t; j < NPTS; j += 256) {
    float dx = pts[3 * j + 0] - px;
    float dy = pts[3 * j + 1] - py;
    float dz = pts[3 * j + 2] - pz;
    float d  = fmaf(dx, dx, fmaf(dy, dy, dz * dz));
    d2[j] = (j == i) ? 3.4e38f : d;
  }
  __syncthreads();
  for (int kk = 0; kk < KNNK; ++kk) {
    float best = 3.4e38f; int bi = 0x7fffffff;
    for (int j = t; j < NPTS; j += 256) {
      float d = d2[j];
      if (d < best || (d == best && j < bi)) { best = d; bi = j; }
    }
    rv[t] = best; ri[t] = bi;
    __syncthreads();
    for (int s = 128; s > 0; s >>= 1) {
      if (t < s) {
        float ov = rv[t + s]; int oi = ri[t + s];
        if (ov < rv[t] || (ov == rv[t] && oi < ri[t])) { rv[t] = ov; ri[t] = oi; }
      }
      __syncthreads();
    }
    if (t == 0) { knn_idx[i * KNNK + kk] = ri[0]; d2[ri[0]] = 3.4e38f; }
    __syncthreads();
  }
}

// ---------------- init: max-feat to enc(-inf), denom to 0 -------------------
__global__ void init_kernel(unsigned* __restrict__ mf, float* __restrict__ denom) {
  int idx = blockIdx.x * blockDim.x + threadIdx.x;
  if (idx < NPTS * HID) mf[idx] = 0x007FFFFFu;   // enc(-inf)
  if (idx < NPTS) denom[idx] = 0.0f;
}

__global__ void zero4_kernel(uint4* __restrict__ p, int n16) {
  int idx = blockIdx.x * blockDim.x + threadIdx.x;
  if (idx < n16) p[idx] = make_uint4(0u, 0u, 0u, 0u);
}

__device__ __forceinline__ void edge_sd(int m, int E, const int* ei,
                                        const int* knn_idx, int& s, int& d) {
  if (m < E) { s = ei[m]; d = ei[E + m]; }
  else { int t = m - E; s = t / KNNK; d = knn_idx[t]; }
}

// ------------- per-edge 3->64 MLP + segment-max via atomicMax ---------------
__global__ void edge_max_kernel(const float* __restrict__ pts, const int* __restrict__ ei,
                                const int* __restrict__ knn_idx,
                                const float* __restrict__ Wt, const float* __restrict__ bt,
                                unsigned* __restrict__ mf, int M, int E) {
  int m = blockIdx.x * blockDim.x + threadIdx.x;
  if (m >= M) return;
  int s, d; edge_sd(m, E, ei, knn_idx, s, d);
  float dx = pts[3 * d + 0] - pts[3 * s + 0];
  float dy = pts[3 * d + 1] - pts[3 * s + 1];
  float dz = pts[3 * d + 2] - pts[3 * s + 2];
  unsigned* row = mf + (size_t)s * HID;
#pragma unroll 4
  for (int f = 0; f < HID; ++f) {
    float h = fmaf(Wt[3 * f + 0], dx, fmaf(Wt[3 * f + 1], dy, fmaf(Wt[3 * f + 2], dz, bt[f])));
    atomicMax(row + f, enc_f32(h));
  }
}

// ------------- dev = relu(max_feat @ W_phi^T + b_phi) -----------------------
__global__ void dev_kernel(const unsigned* __restrict__ mf, const float* __restrict__ W,
                           const float* __restrict__ b, float* __restrict__ dev) {
  int idx = blockIdx.x * blockDim.x + threadIdx.x;
  if (idx >= NPTS * HID) return;
  int i = idx >> 6, f = idx & 63;
  const unsigned* r = mf + (size_t)i * HID;
  float acc = b[f];
#pragma unroll 8
  for (int k = 0; k < HID; ++k) acc = fmaf(dec_f32(r[k]), W[f * HID + k], acc);
  dev[idx] = fmaxf(acc, 0.0f);
}

// ------------- q = dev@Wq^T+bq ; kk = dev@Wk^T+bk ---------------------------
__global__ void qk_kernel(const float* __restrict__ dev,
                          const float* __restrict__ Wq, const float* __restrict__ bq,
                          const float* __restrict__ Wk, const float* __restrict__ bk,
                          float* __restrict__ q, float* __restrict__ kk) {
  int idx = blockIdx.x * blockDim.x + threadIdx.x;
  if (idx >= NPTS * HID) return;
  int i = idx >> 6, f = idx & 63;
  const float* dr = dev + (size_t)i * HID;
  float aq = bq[f], ak = bk[f];
#pragma unroll 8
  for (int k = 0; k < HID; ++k) {
    float dv = dr[k];
    aq = fmaf(dv, Wq[f * HID + k], aq);
    ak = fmaf(dv, Wk[f * HID + k], ak);
  }
  q[idx] = aq; kk[idx] = ak;
}

// ------------- e = exp(q[dst].kk[src]) ; denom = segment_sum ----------------
__global__ void eden_kernel(const int* __restrict__ ei, const int* __restrict__ knn_idx,
                            const float* __restrict__ q, const float* __restrict__ kk,
                            float* __restrict__ e, float* __restrict__ denom, int M, int E) {
  int m = blockIdx.x * blockDim.x + threadIdx.x;
  if (m >= M) return;
  int s, d; edge_sd(m, E, ei, knn_idx, s, d);
  const float* qr = q  + (size_t)d * HID;
  const float* kr = kk + (size_t)s * HID;
  float dot = 0.0f;
#pragma unroll 8
  for (int k = 0; k < HID; ++k) dot = fmaf(qr[k], kr[k], dot);
  float ev = expf(dot);
  e[m] = ev;
  atomicAdd(&denom[s], ev);
}

// ------------- scatter dense f16 attn and A^T -------------------------------
__global__ void scatter_kernel(const int* __restrict__ ei, const int* __restrict__ knn_idx,
                               const float* __restrict__ e, const float* __restrict__ denom,
                               _Float16* __restrict__ attn, _Float16* __restrict__ At,
                               int M, int E) {
  int m = blockIdx.x * blockDim.x + threadIdx.x;
  if (m >= M) return;
  int s, d; edge_sd(m, E, ei, knn_idx, s, d);
  attn[(size_t)s * NPTS + d] = (_Float16)(e[m] / denom[s]);
  At[(size_t)d * NPTS + s]   = (_Float16)1.0f;   // adjacency, stored transposed
}

// ------------- TDM: DMA a [tile1 x tile0] f16 tile into LDS -----------------
// D# layout per CDNA5 ISA (group0 + group1, 2D tensor).
// This toolchain exposes the 6-arg builtin:
//   (uint32x4 g0, int32x8 g1, int32x4, int32x4, int32x8, i32 cpol)
__device__ __forceinline__ void tdm_load_2d(const void* gp, unsigned lds_off,
                                            unsigned dim0, unsigned dim1,
                                            unsigned stride0,
                                            unsigned tile0, unsigned tile1) {
  unsigned long long ga = (unsigned long long)(uintptr_t)gp;
  u32x4 g0;
  g0[0] = 1u;                                               // count=1 (valid), user D#
  g0[1] = lds_off;                                          // lds_addr
  g0[2] = (unsigned)ga;                                     // global_addr[31:0]
  g0[3] = ((unsigned)(ga >> 32) & 0x01FFFFFFu) | 0x80000000u; // ga[56:32] | type=2
  i32x8 g1;
  g1[0] = 0x00010000;                                       // data_size=1 (2 bytes)
  g1[1] = (int)(dim0 << 16);                                // tensor_dim0[15:0]
  g1[2] = (int)((dim0 >> 16) | (dim1 << 16));               // dim0 hi | dim1 lo
  g1[3] = (int)((dim1 >> 16) | (tile0 << 16));              // dim1 hi | tile_dim0
  g1[4] = (int)(tile1 & 0xFFFFu);                           // tile_dim1 (tile_dim2=0)
  g1[5] = (int)stride0;                                     // tensor_dim0_stride lo32
  g1[6] = 0;                                                // stride0 hi | stride1 lo
  g1[7] = 0;                                                // stride1 hi
  i32x4 gz4; gz4[0] = 0; gz4[1] = 0; gz4[2] = 0; gz4[3] = 0; // groups 2/3 unused (2D)
  i32x8 gz8;
  gz8[0] = 0; gz8[1] = 0; gz8[2] = 0; gz8[3] = 0;
  gz8[4] = 0; gz8[5] = 0; gz8[6] = 0; gz8[7] = 0;
  __builtin_amdgcn_tensor_load_to_lds(g0, g1, gz4, gz4, gz8, 0);
}

__device__ __forceinline__ unsigned lds_off_of(const void* p) {
  // Flat shared-aperture address: low 32 bits are the LDS byte offset (ISA sec 10.2).
  return (unsigned)(uintptr_t)p;
}

__device__ __forceinline__ v8f wmma16(v16h a, v16h b, v8f c) {
  return __builtin_amdgcn_wmma_f32_16x16x32_f16(false, a, false, b, (short)0, c, false, false);
}

// ------------- TDM + LDS double-buffered WMMA GEMM: C = A * Bt^T ------------
// A  : [n x kdim] f16 row-major            (WMMA A-operand)
// Bt : [n x kdim] f16 row-major = B^T      (WMMA B-operand)
// Block: 128x128 tile, K-step 64. Wave: 32x64 sub-tile (2x4 fragments).
template <typename OUT_T>
__global__ __launch_bounds__(256) void wmma_gemm_tdm(const _Float16* __restrict__ A,
                                                     const _Float16* __restrict__ Bt,
                                                     OUT_T* __restrict__ C,
                                                     int n, int kdim) {
  __shared__ _Float16 As[2][128][64];
  __shared__ _Float16 Bs[2][128][64];

  const int lane = threadIdx.x & 31;
  const int wave = threadIdx.x >> 5;
  const int wr   = wave >> 1;        // 0..3  -> 32-row band
  const int wc   = wave & 1;         // 0..1  -> 64-col band
  const int mrow = lane & 15;
  const int lhi  = lane >> 4;
  const int rowBase = blockIdx.y * 128;
  const int colBase = blockIdx.x * 128;

  const unsigned asOff = lds_off_of(&As[0][0][0]);
  const unsigned bsOff = lds_off_of(&Bs[0][0][0]);
  const unsigned bufBytes = 128u * 64u * 2u;

  const int nk = kdim / 64;
  if (wave == 0) {
    tdm_load_2d(A  + (size_t)rowBase * kdim, asOff, kdim, n, kdim, 64, 128);
    tdm_load_2d(Bt + (size_t)colBase * kdim, bsOff, kdim, n, kdim, 64, 128);
  }

  v8f acc[2][4] = {};
  for (int s = 0; s < nk; ++s) {
    const int cur = s & 1;
    if (wave == 0) {
      if (s + 1 < nk) {
        const size_t kb = (size_t)(s + 1) * 64;
        tdm_load_2d(A  + (size_t)rowBase * kdim + kb, asOff + (1 - cur) * bufBytes,
                    kdim, n, kdim, 64, 128);
        tdm_load_2d(Bt + (size_t)colBase * kdim + kb, bsOff + (1 - cur) * bufBytes,
                    kdim, n, kdim, 64, 128);
        __builtin_amdgcn_s_wait_tensorcnt(2);   // current pair complete
      } else {
        __builtin_amdgcn_s_wait_tensorcnt(0);
      }
    }
    __syncthreads();                            // tile ready for all waves

#pragma unroll
    for (int kk = 0; kk < 64; kk += 32) {
      // A fragments for the wave's two 16-row tiles
      const _Float16* ar0 = &As[cur][wr * 32 + mrow][kk + lhi * 8];
      const _Float16* ar1 = &As[cur][wr * 32 + 16 + mrow][kk + lhi * 8];
      v8h a0lo = *(const v8h*)ar0;
      v8h a0hi = *(const v8h*)(ar0 + 16);
      v8h a1lo = *(const v8h*)ar1;
      v8h a1hi = *(const v8h*)(ar1 + 16);
      v16h a0 = __builtin_shufflevector(a0lo, a0hi, 0, 1, 2, 3, 4, 5, 6, 7,
                                                   8, 9, 10, 11, 12, 13, 14, 15);
      v16h a1 = __builtin_shufflevector(a1lo, a1hi, 0, 1, 2, 3, 4, 5, 6, 7,
                                                   8, 9, 10, 11, 12, 13, 14, 15);
#pragma unroll
      for (int ct = 0; ct < 4; ++ct) {
        const _Float16* br = &Bs[cur][wc * 64 + ct * 16 + mrow][kk + lhi * 8];
        v16h b = __builtin_shufflevector(*(const v8h*)br, *(const v8h*)(br + 16), 0, 1, 2, 3, 4, 5, 6, 7,
                                         8, 9, 10, 11, 12, 13, 14, 15);
        acc[0][ct] = wmma16(a0, b, acc[0][ct]);
        acc[1][ct] = wmma16(a1, b, acc[1][ct]);
      }
    }
    __syncthreads();                            // done with buffer before DMA reuse
  }

  // C/D layout: VGPR r -> M = r + lhi*8 ; N = lane&15
#pragma unroll
  for (int rt = 0; rt < 2; ++rt) {
#pragma unroll
    for (int r = 0; r < 8; ++r) {
      size_t row = (size_t)(rowBase + wr * 32 + rt * 16 + lhi * 8 + r);
      size_t o   = row * n + colBase + wc * 64 + mrow;
      C[o +  0] = (OUT_T)acc[rt][0][r];
      C[o + 16] = (OUT_T)acc[rt][1][r];
      C[o + 32] = (OUT_T)acc[rt][2][r];
      C[o + 48] = (OUT_T)acc[rt][3][r];
    }
  }
}

extern "C" void kernel_launch(void* const* d_in, const int* in_sizes, int n_in,
                              void* d_out, int out_size, void* d_ws, size_t ws_size,
                              hipStream_t stream) {
  const float* points = (const float*)d_in[0];
  // d_in[1] = features (unused by reference)
  const int*   ei     = (const int*)d_in[2];
  const float* Wt     = (const float*)d_in[3];
  const float* bt     = (const float*)d_in[4];
  const float* Wphi   = (const float*)d_in[5];
  const float* bphi   = (const float*)d_in[6];
  const float* Wq     = (const float*)d_in[7];
  const float* bq     = (const float*)d_in[8];
  const float* Wk     = (const float*)d_in[9];
  const float* bk     = (const float*)d_in[10];
  float* out = (float*)d_out;

  const int E = in_sizes[2] / 2;
  const int M = E + NPTS * KNNK;

  // ---- workspace layout (bytes) ----
  char* ws = (char*)d_ws;
  _Float16* attn_h = (_Float16*)(ws + 0);                  // 33,554,432
  _Float16* At_h   = (_Float16*)(ws + 33554432);           // 33,554,432
  _Float16* B1_h   = (_Float16*)(ws + 67108864);           // 33,554,432
  unsigned* mf     = (unsigned*)(ws + 100663296);          //  1,048,576
  float*    devb   = (float*)   (ws + 101711872);          //  1,048,576
  float*    qb     = (float*)   (ws + 102760448);          //  1,048,576
  float*    kkb    = (float*)   (ws + 103809024);          //  1,048,576
  float*    eb     = (float*)   (ws + 104857600);          //    393,216 (pad)
  float*    denom  = (float*)   (ws + 105250816);          //     16,384
  int*      knn    = (int*)     (ws + 105267200);          //    245,760

  const int TPB = 256;
  const int egrid = (M + TPB - 1) / TPB;

  // 1) KNN neighbor lists
  knn_kernel<<<NPTS, TPB, 0, stream>>>(points, knn);
  // 2) init segment-max accumulators + denom
  init_kernel<<<(NPTS * HID + TPB - 1) / TPB, TPB, 0, stream>>>(mf, denom);
  // 3) zero dense attn + A^T (contiguous 64 MB, 16B vector stores)
  zero4_kernel<<<(67108864 / 16 + TPB - 1) / TPB, TPB, 0, stream>>>((uint4*)attn_h, 67108864 / 16);
  // 4) per-edge theta MLP + segment max
  edge_max_kernel<<<egrid, TPB, 0, stream>>>(points, ei, knn, Wt, bt, mf, M, E);
  // 5) dev = relu(max_feat @ Wphi^T + bphi)
  dev_kernel<<<(NPTS * HID + TPB - 1) / TPB, TPB, 0, stream>>>(mf, Wphi, bphi, devb);
  // 6) q, k projections
  qk_kernel<<<(NPTS * HID + TPB - 1) / TPB, TPB, 0, stream>>>(devb, Wq, bq, Wk, bk, qb, kkb);
  // 7) edge softmax numerators + denominators
  eden_kernel<<<egrid, TPB, 0, stream>>>(ei, knn, qb, kkb, eb, denom, M, E);
  // 8) scatter dense f16 attn and A^T
  scatter_kernel<<<egrid, TPB, 0, stream>>>(ei, knn, eb, denom, attn_h, At_h, M, E);
  // 9) B1 = attn @ A   (Bt = A^T row-major), f16 out
  dim3 ggrid(NPTS / 128, NPTS / 128);
  wmma_gemm_tdm<_Float16><<<ggrid, TPB, 0, stream>>>(attn_h, At_h, B1_h, NPTS, NPTS);
  // 10) A_s = B1 @ attn^T  (Bt = attn row-major), f32 out
  wmma_gemm_tdm<float><<<ggrid, TPB, 0, stream>>>(B1_h, attn_h, out, NPTS, NPTS);
}
